// HibCriterion_1743756722653
// MI455X (gfx1250) — matrix operand order repaired and checked by reading, not compiled, not run on hardware
//
#include <hip/hip_runtime.h>
#include <hip/hip_bf16.h>

// ---------------------------------------------------------------------------
// HIB criterion for MI455X (gfx1250, wave32).
//
// d2[i][j] = ||xa_i||^2 + ||xb_j||^2 - 2 <xa_i, xb_j>
// The 8x128 @ 128x8 inner-product block is done with V_WMMA_F32_16X16X32_F16,
// packing TWO pairs per 16x16 tile (diagonal 8x8 blocks are the useful data).
// Norms are accumulated in f32 from the original data during the loads.
// 64 MB z-table lives in L2 (192 MB); kernel is L2-gather bound.
// ---------------------------------------------------------------------------

typedef __attribute__((ext_vector_type(16))) _Float16 v16h;
typedef __attribute__((ext_vector_type(8)))  float    v8f;

#define HIB_EPS   1e-6f
#define NBLK      1024      // partial-sum slots in d_ws (4 KB)
#define BLOCK     256       // 8 waves of 32

__device__ __forceinline__ float sq4(float4 v) {
    return v.x * v.x + v.y * v.y + v.z * v.z + v.w * v.w;
}

// One 16x16 tile: A = rows of z[ia0](m<8) / z[ia1](m>=8), B^T likewise from
// ib0/ib1.  Returns f32 accumulator G = A.B^T (v8f, D-matrix layout) plus the
// full f32 row norms na[m=lane&15], nb[n=lane&15] (combined across hi-halves).
__device__ __forceinline__ v8f tile_gemm(const float* __restrict__ z,
                                         int ia0, int ia1, int ib0, int ib1,
                                         int lane, float& naOut, float& nbOut)
{
    const int m  = lane & 15;
    const int hi = lane >> 4;
    const float4* rA = reinterpret_cast<const float4*>(
        z + (size_t)(m < 8 ? ia0 : ia1) * 1024u + (size_t)(m & 7) * 128u);
    const float4* rB = reinterpret_cast<const float4*>(
        z + (size_t)(m < 8 ? ib0 : ib1) * 1024u + (size_t)(m & 7) * 128u);

    v8f acc = {0.f, 0.f, 0.f, 0.f, 0.f, 0.f, 0.f, 0.f};
    float na = 0.f, nb = 0.f;

#pragma unroll
    for (int b = 0; b < 4; ++b) {           // 4 K-blocks of 32
        // A lane data: K = b*32 + {hi*8+0..7, 16+hi*8+0..7}
        float4 a0 = rA[b * 8 + hi * 2 + 0];
        float4 a1 = rA[b * 8 + hi * 2 + 1];
        float4 a2 = rA[b * 8 + hi * 2 + 4];
        float4 a3 = rA[b * 8 + hi * 2 + 5];
        // B lane data: K = b*32 + {hi*8+0..7, 16+hi*8+0..7}
        float4 b0 = rB[b * 8 + hi * 2 + 0];
        float4 b1 = rB[b * 8 + hi * 2 + 1];
        float4 b2 = rB[b * 8 + hi * 2 + 4];
        float4 b3 = rB[b * 8 + hi * 2 + 5];

        na += sq4(a0) + sq4(a1) + sq4(a2) + sq4(a3);
        nb += sq4(b0) + sq4(b1) + sq4(b2) + sq4(b3);

        v16h av, bv;
        av[0]  = (_Float16)a0.x; av[1]  = (_Float16)a0.y;
        av[2]  = (_Float16)a0.z; av[3]  = (_Float16)a0.w;
        av[4]  = (_Float16)a1.x; av[5]  = (_Float16)a1.y;
        av[6]  = (_Float16)a1.z; av[7]  = (_Float16)a1.w;
        av[8]  = (_Float16)a2.x; av[9]  = (_Float16)a2.y;
        av[10] = (_Float16)a2.z; av[11] = (_Float16)a2.w;
        av[12] = (_Float16)a3.x; av[13] = (_Float16)a3.y;
        av[14] = (_Float16)a3.z; av[15] = (_Float16)a3.w;

        bv[0]  = (_Float16)b0.x; bv[1]  = (_Float16)b0.y;
        bv[2]  = (_Float16)b0.z; bv[3]  = (_Float16)b0.w;
        bv[4]  = (_Float16)b1.x; bv[5]  = (_Float16)b1.y;
        bv[6]  = (_Float16)b1.z; bv[7]  = (_Float16)b1.w;
        bv[8]  = (_Float16)b2.x; bv[9]  = (_Float16)b2.y;
        bv[10] = (_Float16)b2.z; bv[11] = (_Float16)b2.w;
        bv[12] = (_Float16)b3.x; bv[13] = (_Float16)b3.y;
        bv[14] = (_Float16)b3.z; bv[15] = (_Float16)b3.w;

        acc = __builtin_amdgcn_wmma_f32_16x16x32_f16(
            /*neg_a=*/false, av, /*neg_b=*/false, bv,
            /*c_mod=*/(short)0, acc, /*reuse_a=*/false, /*reuse_b=*/false);
    }
    // combine K-halves held by lane and lane^16 (wave32 shuffle)
    naOut = na + __shfl_xor(na, 16, 32);
    nbOut = nb + __shfl_xor(nb, 16, 32);
    return acc;
}

// Loss over the two valid diagonal 8x8 blocks of the tile.
__device__ __forceinline__ float tile_loss(v8f g, float na_comb, float nb_comb,
                                           float a, float beta, int lane,
                                           bool isPos, bool pair1ok)
{
    const int hi = lane >> 4;
    float sum = 0.f;
#pragma unroll
    for (int r = 0; r < 8; ++r) {
        // M = r + 8*hi ; na for row M lives in lane (r + 8*hi)
        float naM = __shfl(na_comb, hi * 8 + r, 32);
        float d2  = naM + nb_comb - 2.0f * g[r];
        float t   = fmaf(-a, d2, beta);
        float s   = 1.0f / (1.0f + __expf(-t));          // sigmoid(t)
        float term = isPos ? -__logf(s + HIB_EPS)
                           : -__logf(1.0f - s - HIB_EPS);
        sum += term;
    }
    // valid: (M<8 && N<8)  -> lanes 0..7   (pair0 block)
    //        (M>=8 && N>=8)-> lanes 24..31 (pair1 block, if it exists)
    bool valid = (lane < 8) || (lane >= 24 && pair1ok);
    return valid ? sum : 0.f;
}

__global__ void __launch_bounds__(BLOCK)
hib_main_kernel(const float* __restrict__ z,
                const float* __restrict__ alphaP,
                const float* __restrict__ betaP,
                const int* __restrict__ ap, const int* __restrict__ p,
                const int* __restrict__ an, const int* __restrict__ n,
                float* __restrict__ partials, int P)
{
    const int lane  = threadIdx.x & 31;
    const int wave  = threadIdx.x >> 5;
    const int gwave = blockIdx.x * (BLOCK / 32) + wave;
    const int nwav  = gridDim.x * (BLOCK / 32);
    const int units = (P + 1) >> 1;          // 2 pairs per 16x16 tile

    const float aSP  = log1pf(__expf(alphaP[0]));  // softplus(alpha)
    const float beta = betaP[0];

    float acc = 0.f;
    for (int u = gwave; u < units; u += nwav) {      // wave-uniform loop
        const int k0 = 2 * u, k1 = 2 * u + 1;
        const bool p1ok = (k1 < P);
        const int k1c = p1ok ? k1 : k0;

        const int iap0 = ap[k0], iap1 = ap[k1c];
        const int ip0  = p[k0],  ip1  = p[k1c];
        const int ian0 = an[k0], ian1 = an[k1c];
        const int in0  = n[k0],  in1  = n[k1c];

        // Prefetch next unit's gathered 4 KB row-blocks (one 128 B line/lane)
        const int un = u + nwav;
        if (un < units) {
            const int kn = 2 * un;
            __builtin_prefetch(z + (size_t)ap[kn] * 1024u + lane * 32, 0, 1);
            __builtin_prefetch(z + (size_t)p[kn]  * 1024u + lane * 32, 0, 1);
            __builtin_prefetch(z + (size_t)an[kn] * 1024u + lane * 32, 0, 1);
            __builtin_prefetch(z + (size_t)n[kn]  * 1024u + lane * 32, 0, 1);
        }

        float na, nb;
        v8f gp = tile_gemm(z, iap0, iap1, ip0, ip1, lane, na, nb);
        acc += tile_loss(gp, na, nb, aSP, beta, lane, /*isPos=*/true,  p1ok);
        v8f gn = tile_gemm(z, ian0, ian1, in0, in1, lane, na, nb);
        acc += tile_loss(gn, na, nb, aSP, beta, lane, /*isPos=*/false, p1ok);
    }

    // wave32 reduce, then block reduce -> deterministic per-block partial
#pragma unroll
    for (int off = 16; off >= 1; off >>= 1)
        acc += __shfl_xor(acc, off, 32);

    __shared__ float smem[BLOCK / 32];
    if (lane == 0) smem[wave] = acc;
    __syncthreads();
    if (threadIdx.x == 0) {
        float s = 0.f;
        for (int i = 0; i < BLOCK / 32; ++i) s += smem[i];
        partials[blockIdx.x] = s;
    }
}

__global__ void __launch_bounds__(BLOCK)
hib_reduce_kernel(const float* __restrict__ partials, int nparts,
                  float* __restrict__ out, int P)
{
    float s = 0.f;
    for (int i = threadIdx.x; i < nparts; i += BLOCK) s += partials[i];
#pragma unroll
    for (int off = 16; off >= 1; off >>= 1)
        s += __shfl_xor(s, off, 32);

    __shared__ float smem[BLOCK / 32];
    const int lane = threadIdx.x & 31, wave = threadIdx.x >> 5;
    if (lane == 0) smem[wave] = s;
    __syncthreads();
    if (threadIdx.x == 0) {
        float t = 0.f;
        for (int i = 0; i < BLOCK / 32; ++i) t += smem[i];
        out[0] = t / (64.0f * (float)P);    // /(S*S) and mean over P
    }
}

extern "C" void kernel_launch(void* const* d_in, const int* in_sizes, int n_in,
                              void* d_out, int out_size, void* d_ws, size_t ws_size,
                              hipStream_t stream)
{
    const float* z      = (const float*)d_in[0];   // [16384, 8, 128] f32
    const float* alphaP = (const float*)d_in[1];   // scalar
    const float* betaP  = (const float*)d_in[2];   // scalar
    const int*   ap     = (const int*)d_in[3];     // [P]
    const int*   p      = (const int*)d_in[4];
    const int*   an     = (const int*)d_in[5];
    const int*   n      = (const int*)d_in[6];
    const int P = in_sizes[3];

    float* partials = (float*)d_ws;                // NBLK floats (4 KB)
    float* out      = (float*)d_out;

    hib_main_kernel<<<NBLK, BLOCK, 0, stream>>>(z, alphaP, betaP,
                                                ap, p, an, n, partials, P);
    hib_reduce_kernel<<<1, BLOCK, 0, stream>>>(partials, NBLK, out, P);
}
